// QMeasureDensityMPS_85117661872522
// MI455X (gfx1250) — matrix operand order repaired and checked by reading, not compiled, not run on hardware
//
#include <hip/hip_runtime.h>
#include <hip/hip_bf16.h>

typedef __attribute__((ext_vector_type(16))) _Float16 v16h;
typedef __attribute__((ext_vector_type(8)))  float    v8f;
typedef __attribute__((ext_vector_type(2)))  float    v2f;

// Problem constants
#define NB      65536   // batch
#define ROWLEN  512     // floats per batch row of x (256 sites * 2 phys)
#define KPAD    512     // padded K (= 2*254 -> 512)
#define NCHUNK  16      // KPAD / 32
#define DB      64      // bond dim / N
// Packed weight buffer: [half(2)][chunk(16)][ntile(4)][lane(32)][16 halves]
// = 2*16*4*32*16 f16 = 128 KB, lives in d_ws (stays hot in L2/WGP$).

// ---------------------------------------------------------------------------
// Prep: W2[j,c] = sum_b mps0[b,p0] * mps1[s,b,c,p],  j = 2*t + p0, t = 2*s + p
// written directly in WMMA B-fragment order (16-bit B 32x16 per ISA 7.12.2).
// ---------------------------------------------------------------------------
__global__ void mps_prep_weights(const float* __restrict__ mps0,
                                 const float* __restrict__ mps1,
                                 const float* __restrict__ mps2,
                                 const float* __restrict__ mps3,
                                 _Float16* __restrict__ wpk) {
    int idx = blockIdx.x * blockDim.x + threadIdx.x;   // 0 .. 65535
    int h = idx >> 15;            // 0 = left, 1 = right
    int r = idx & 32767;
    int j = r >> 6;               // 0 .. 511  (K index)
    int c = r & 63;               // 0 .. 63   (N index)

    int p0 = j & 1;
    int t  = j >> 1;              // 0 .. 255, valid < 254
    float val = 0.0f;
    if (t < 254) {
        int s = t >> 1;
        int p = t & 1;
        const float* m0 = h ? mps2 : mps0;
        const float* m1 = h ? mps3 : mps1;
        // mps1[s,b,c,p] flat: ((s*64 + b)*64 + c)*2 + p ; b-stride = 128 floats
        const float* mrow = m1 + ((size_t)s * 64 * 64 + c) * 2 + p;
        #pragma unroll 8
        for (int b = 0; b < 64; ++b)
            val += m0[b * 2 + p0] * mrow[(size_t)b * 128];
    }

    // Fragment placement (ISA: B-matrix 16-bit 32x16):
    //   lanes 0-15 : N = lane,    elements 0..7 -> K 0..7,  8..15 -> K 16..23
    //   lanes 16-31: N = lane-16, elements 0..7 -> K 8..15, 8..15 -> K 24..31
    int q    = j >> 5;            // chunk
    int kl   = j & 31;            // K within chunk
    int lane = (((kl >> 3) & 1) ? 16 : 0) | (c & 15);
    int e    = (kl & 7) | ((kl >> 4) << 3);
    int n    = c >> 4;            // N-tile
    size_t pos = ((((size_t)(h * 16 + q) * 4 + n) * 32 + lane) * 16) + e;
    wpk[pos] = (_Float16)val;
}

// ---------------------------------------------------------------------------
// Non-temporal x loader: 8 floats (t = +0..3 and +8..11) for one lane-chunk.
// x is streamed exactly once; NT keeps L2 reserved for the weights.
// ---------------------------------------------------------------------------
__device__ __forceinline__ void load_x8_nt(const float* __restrict__ p,
                                           float* __restrict__ v) {
    v2f a = __builtin_nontemporal_load((const v2f*)(p));
    v2f b = __builtin_nontemporal_load((const v2f*)(p + 2));
    v2f c = __builtin_nontemporal_load((const v2f*)(p + 8));
    v2f d = __builtin_nontemporal_load((const v2f*)(p + 10));
    v[0] = a[0]; v[1] = a[1]; v[2] = b[0]; v[3] = b[1];
    v[4] = c[0]; v[5] = c[1]; v[6] = d[0]; v[7] = d[1];
}

// ---------------------------------------------------------------------------
// A-fragment generator: 16-bit A 16x32 layout (ISA 7.12.2):
//  lanes 0-15 : M=lane,    elements 0..7 -> K 0..7,  8..15 -> K 16..23
//  lanes 16-31: M=lane-16, elements 0..7 -> K 8..15, 8..15 -> K 24..31
// With j = 2t+p0 each lane needs t = tbase+{0..3} and tbase+{8..11}, each
// multiplied by the two boundary scalars (p0 = 0,1).
// ---------------------------------------------------------------------------
__device__ __forceinline__ v16h make_afrag(const float* __restrict__ v,
                                           float s0, float s1) {
    v16h A;
    #pragma unroll
    for (int i = 0; i < 4; ++i) {
        A[2 * i]         = (_Float16)(v[i] * s0);
        A[2 * i + 1]     = (_Float16)(v[i] * s1);
        A[8 + 2 * i]     = (_Float16)(v[4 + i] * s0);
        A[8 + 2 * i + 1] = (_Float16)(v[4 + i] * s1);
    }
    return A;
}

// ---------------------------------------------------------------------------
// Main: out[B] = sum_c left[B,c] * right[B,c]
// left  = f_left  @ W2,  f_left [B, 2t+p0] = x[B,0..1][p0]     * xb[2   + t]
// right = f_right @ W3,  f_right[B, 2t+p0] = x[B,510..511][p0] * xb[256 + t]
// One 16-row tile per wave, 8 waves per block; x double-buffered a chunk
// ahead; weight fragments loaded as a grouped clause per chunk.
// ---------------------------------------------------------------------------
__global__ void __launch_bounds__(256)
mps_measure_main(const float* __restrict__ x,
                 const _Float16* __restrict__ wpk,
                 float* __restrict__ out) {
    const int lane = threadIdx.x & 31;
    const int wave = threadIdx.x >> 5;
    const int tile = blockIdx.x * 8 + wave;       // 0 .. 4095
    const int rowBase = tile * 16;

    const int m   = lane & 15;                    // row within tile
    const int thi = (lane >> 4) & 1;              // lane-half selector
    const float* xb = x + (size_t)(rowBase + m) * ROWLEN;

    // boundary scalars for this row
    const float s0L = xb[0],   s1L = xb[1];
    const float s0R = xb[510], s1R = xb[511];

    v8f accL[4] = {v8f{}, v8f{}, v8f{}, v8f{}};
    v8f accR[4] = {v8f{}, v8f{}, v8f{}, v8f{}};

    const _Float16* wL = wpk;                               // half 0
    const _Float16* wR = wpk + (size_t)16 * 4 * 32 * 16;    // half 1

    // x double buffers (registers after unroll-by-2)
    float xLb[2][8], xRb[2][8];
    {
        const int tb0 = thi * 4;
        load_x8_nt(xb + 2   + tb0, xLb[0]);
        load_x8_nt(xb + 256 + tb0, xRb[0]);
    }

    #pragma unroll 2
    for (int q = 0; q < NCHUNK; ++q) {
        const int cur = q & 1;
        const int nxt = cur ^ 1;

        // (1) grouped weight-fragment loads for this chunk (8 live values)
        v16h BL[4], BR[4];
        #pragma unroll
        for (int n = 0; n < 4; ++n) {
            BL[n] = *(const v16h*)(wL + ((size_t)(q * 4 + n) * 32 + lane) * 16);
            BR[n] = *(const v16h*)(wR + ((size_t)(q * 4 + n) * 32 + lane) * 16);
        }

        // (2) prefetch next chunk's x stream (non-temporal)
        if (q < NCHUNK - 1) {
            const int tb = (q + 1) * 16 + thi * 4;
            load_x8_nt(xb + 2   + tb, xLb[nxt]);
            load_x8_nt(xb + 256 + tb, xRb[nxt]);
        }

#if __has_builtin(__builtin_amdgcn_sched_barrier)
        // Keep all loads hoisted above the VALU/WMMA block so waits pipeline.
        __builtin_amdgcn_sched_barrier(0);
#endif

        // (3) build A fragments (pure VALU — covers WGP$ latency of (1))
        v16h AL = make_afrag(xLb[cur], s0L, s1L);
        v16h AR = make_afrag(xRb[cur], s0R, s1R);

        // (4) 8 WMMAs for this chunk
        #pragma unroll
        for (int n = 0; n < 4; ++n) {
            accL[n] = __builtin_amdgcn_wmma_f32_16x16x32_f16(
                false, AL, false, BL[n], (short)0, accL[n], false, false);
            accR[n] = __builtin_amdgcn_wmma_f32_16x16x32_f16(
                false, AR, false, BR[n], (short)0, accR[n], false, false);
        }
    }

    // C/D layout: lanes 0-15: M = r; lanes 16-31: M = r + 8; N = (lane&15)+16n.
    // out[M] = sum over all 64 columns -> per-lane partial dot, then 16-lane
    // butterfly reduction within each lane half.
    #pragma unroll
    for (int r = 0; r < 8; ++r) {
        float v = 0.0f;
        #pragma unroll
        for (int n = 0; n < 4; ++n)
            v += accL[n][r] * accR[n][r];
        v += __shfl_xor(v, 1, 16);
        v += __shfl_xor(v, 2, 16);
        v += __shfl_xor(v, 4, 16);
        v += __shfl_xor(v, 8, 16);
        if ((lane & 15) == 0)
            out[rowBase + thi * 8 + r] = v;
    }
}

// ---------------------------------------------------------------------------
extern "C" void kernel_launch(void* const* d_in, const int* in_sizes, int n_in,
                              void* d_out, int out_size, void* d_ws, size_t ws_size,
                              hipStream_t stream) {
    const float* x    = (const float*)d_in[0];
    const float* mps0 = (const float*)d_in[1];
    const float* mps1 = (const float*)d_in[2];
    const float* mps2 = (const float*)d_in[3];
    const float* mps3 = (const float*)d_in[4];
    float* out        = (float*)d_out;
    _Float16* wpk     = (_Float16*)d_ws;   // 128 KB packed weights

    (void)in_sizes; (void)n_in; (void)out_size; (void)ws_size;

    // Build fragment-packed contracted weights (2*512*64 outputs).
    mps_prep_weights<<<256, 256, 0, stream>>>(mps0, mps1, mps2, mps3, wpk);

    // 4096 tiles of 16 rows, 8 waves per block.
    mps_measure_main<<<512, 256, 0, stream>>>(x, wpk, out);
}
